// Acoustic_39900246180054
// MI455X (gfx1250) — matrix operand order repaired and checked by reading, not compiled, not run on hardware
//
#include <hip/hip_runtime.h>
#include <hip/hip_bf16.h>

typedef __attribute__((ext_vector_type(16))) _Float16 v16h;
typedef __attribute__((ext_vector_type(8)))  _Float16 v8hf;
typedef __attribute__((ext_vector_type(8)))  float    v8f;

// ---------------------------------------------------------------------------
// WMMA helpers (CDNA5 wave32, V_WMMA_F32_16X16X32_F16)
// ---------------------------------------------------------------------------
__device__ __forceinline__ v8f wmma_f16(v16h a, v16h b, v8f c) {
    // (neg_a, A, neg_b, B, c_mod, C, reuse_a, reuse_b)
    return __builtin_amdgcn_wmma_f32_16x16x32_f16(false, a, false, b, (short)0, c,
                                                  false, false);
}

// A-matrix fragment (16x32 f16, row-major X with leading dim ld):
// lanes 0-15 hold row (lane&15), K = {k0..k0+7, k0+16..k0+23}
// lanes 16-31 hold K = {k0+8..k0+15, k0+24..k0+31}
__device__ __forceinline__ v16h load_frag_a(const _Float16* X, int ld, int m0,
                                            int k0, int lane) {
    int row = m0 + (lane & 15);
    int k   = k0 + ((lane >> 4) << 3);
    const _Float16* p = X + (size_t)row * ld + k;
    v8hf lo = *(const v8hf*)(p);
    v8hf hi = *(const v8hf*)(p + 16);
    v16h f;
#pragma unroll
    for (int i = 0; i < 8; ++i) { f[i] = lo[i]; f[i + 8] = hi[i]; }
    return f;
}

// B-matrix fragment (32x16 f16). B = W^T where W is (N x K) row-major, so
// column n of B is row n of W. Lane l holds column (l&15),
// K = {k0..k0+7, k0+16..k0+23} for lanes 0-15, {k0+8..k0+15, k0+24..k0+31} for lanes 16-31.
__device__ __forceinline__ v16h load_frag_b(const _Float16* W, int ld, int n0,
                                            int k0, int lane) {
    int rowW = n0 + (lane & 15);
    int k    = k0 + ((lane >> 4) << 3);
    const _Float16* p = W + (size_t)rowW * ld + k;
    v8hf lo = *(const v8hf*)(p);
    v8hf hi = *(const v8hf*)(p + 16);
    v16h f;
#pragma unroll
    for (int i = 0; i < 8; ++i) { f[i] = lo[i]; f[i + 8] = hi[i]; }
    return f;
}

__device__ __forceinline__ float sigmoidf(float x) {
    return 1.0f / (1.0f + __expf(-x));
}

// ---------------------------------------------------------------------------
// Small utility kernels
// ---------------------------------------------------------------------------
__global__ void f32_to_f16_kernel(const float* __restrict__ s,
                                  _Float16* __restrict__ d, int n) {
    int i = blockIdx.x * blockDim.x + threadIdx.x;
    if (i < n) d[i] = (_Float16)s[i];
}

__global__ void fill_zero_h_kernel(_Float16* __restrict__ p, int n) {
    int i = blockIdx.x * blockDim.x + threadIdx.x;
    if (i < n) p[i] = (_Float16)0.0f;
}

// Gather note embedding rows into f16 activation matrix (M x 256).
__global__ void embed_gather_kernel(const int* __restrict__ note,
                                    const float* __restrict__ emb,
                                    _Float16* __restrict__ ne, int M) {
    int idx = blockIdx.x * blockDim.x + threadIdx.x;
    if (idx >= M * 256) return;
    int m = idx >> 8, k = idx & 255;
    int id = note[m] & 127;  // NOTE=128
    ne[idx] = (_Float16)emb[id * 256 + k];
}

// Assemble decoder input (T*B, 1088): cols 0..1023 = enc[b*T+t], cols
// 1024..1087 = prev[b,t]*W[k]+bias[k]. Row m = t*32 + b (scan-major layout).
__global__ void build_decin_kernel(const _Float16* __restrict__ enc,
                                   const float* __restrict__ prev,
                                   const float* __restrict__ W,
                                   const float* __restrict__ bias,
                                   _Float16* __restrict__ decin) {
    long long idx = (long long)blockIdx.x * blockDim.x + threadIdx.x;
    const long long total = 25600LL * 1088LL;
    if (idx >= total) return;
    int m = (int)(idx / 1088);
    int c = (int)(idx % 1088);
    int t = m >> 5, b = m & 31;
    if (c < 1024) {
        decin[idx] = enc[(size_t)(b * 800 + t) * 1024 + c];
    } else {
        int k = c - 1024;
        decin[idx] = (_Float16)(prev[b * 800 + t] * W[k] + bias[k]);
    }
}

// ---------------------------------------------------------------------------
// Generic WMMA GEMM: Y = act(X @ W^T + bias)
//   X: (M x K) f16 row-major, W: (N x K) f16 row-major, bias: f32[N]
//   Y: (M x N) f16. One wave computes a 32x64 output macro-tile
//   (2 row-tiles x 4 col-tiles): each B fragment is reused by 2 WMMAs and
//   each A fragment by 4 -> 12 b128 loads per 8 WMMAs. Block = 4 waves.
// ---------------------------------------------------------------------------
template <int ACT>  // 0 = none, 1 = tanh
__global__ void gemm_bias_act_kernel(const _Float16* __restrict__ X,
                                     const _Float16* __restrict__ W,
                                     const float* __restrict__ bias,
                                     _Float16* __restrict__ Y,
                                     int M, int N, int K) {
    int lane = threadIdx.x & 31;
    int wave = threadIdx.x >> 5;
    int tile = blockIdx.x * 4 + wave;
    int tiles_n = N >> 6;
    int m0  = (tile / tiles_n) << 5;   // 32-row macro tile
    int n0g = (tile % tiles_n) << 6;   // 64-col macro tile

    v8f acc[2][4];
#pragma unroll
    for (int i = 0; i < 2; ++i)
#pragma unroll
        for (int j = 0; j < 4; ++j)
#pragma unroll
            for (int r = 0; r < 8; ++r) acc[i][j][r] = 0.0f;

    for (int k = 0; k < K; k += 32) {
        v16h a0 = load_frag_a(X, K, m0, k, lane);
        v16h a1 = load_frag_a(X, K, m0 + 16, k, lane);
#pragma unroll
        for (int j = 0; j < 4; ++j) {
            v16h b = load_frag_b(W, K, n0g + 16 * j, k, lane);
            acc[0][j] = wmma_f16(a0, b, acc[0][j]);
            acc[1][j] = wmma_f16(a1, b, acc[1][j]);
        }
    }

    int col = lane & 15, rh = lane >> 4;
#pragma unroll
    for (int j = 0; j < 4; ++j) {
        int n = n0g + 16 * j + col;
        float bv = bias[n];
#pragma unroll
        for (int i = 0; i < 2; ++i) {
#pragma unroll
            for (int r = 0; r < 8; ++r) {
                int row = m0 + 16 * i + r + (rh << 3);
                float v = acc[i][j][r] + bv;
                if (ACT == 1) v = tanhf(v);
                Y[(size_t)row * N + n] = (_Float16)v;
            }
        }
    }
}

// ---------------------------------------------------------------------------
// Encoder GRU step (scan over B, state = 800 x 512). One wave computes a
// 32x16 h macro-tile: 2 row-tiles sharing the three fused gate B-fragments
// (r/z/n columns of gh = h@Whh^T + bhh). gi already contains x@Wih^T + bih.
// Grid: (800/32)*(512/16) = 800 waves -> 200 blocks of 128.
// ---------------------------------------------------------------------------
__global__ void gru_enc_step_kernel(const _Float16* __restrict__ gi,    // 800 x 1536
                                    const _Float16* __restrict__ Whh,   // 1536 x 512
                                    const float* __restrict__ bhh,      // 1536
                                    const _Float16* __restrict__ hprev, // 800 x 512
                                    _Float16* __restrict__ hnext,       // 800 x 512
                                    _Float16* __restrict__ encout)      // rows 800, ld 1024
{
    const int H = 512;
    int lane = threadIdx.x & 31;
    int wave = threadIdx.x >> 5;
    int tile = blockIdx.x * 4 + wave;   // 0..799
    int m0 = (tile >> 5) << 5;          // 25 row groups of 32
    int n0 = (tile & 31) << 4;          // 32 col tiles
    int col = lane & 15, rh = lane >> 4;

    float br = bhh[n0 + col];
    float bz = bhh[512 + n0 + col];
    float bn = bhh[1024 + n0 + col];
    v8f ar[2], az[2], an[2];
#pragma unroll
    for (int i = 0; i < 2; ++i)
#pragma unroll
        for (int r = 0; r < 8; ++r) { ar[i][r] = br; az[i][r] = bz; an[i][r] = bn; }

    for (int k = 0; k < H; k += 32) {
        v16h a0 = load_frag_a(hprev, H, m0, k, lane);
        v16h a1 = load_frag_a(hprev, H, m0 + 16, k, lane);
        v16h b0 = load_frag_b(Whh, H, n0, k, lane);
        v16h b1 = load_frag_b(Whh, H, 512 + n0, k, lane);
        v16h b2 = load_frag_b(Whh, H, 1024 + n0, k, lane);
        ar[0] = wmma_f16(a0, b0, ar[0]);
        az[0] = wmma_f16(a0, b1, az[0]);
        an[0] = wmma_f16(a0, b2, an[0]);
        ar[1] = wmma_f16(a1, b0, ar[1]);
        az[1] = wmma_f16(a1, b1, az[1]);
        an[1] = wmma_f16(a1, b2, an[1]);
    }

#pragma unroll
    for (int i = 0; i < 2; ++i) {
#pragma unroll
        for (int r = 0; r < 8; ++r) {
            int row = m0 + 16 * i + r + (rh << 3);
            const _Float16* gir = gi + (size_t)row * 1536;
            float ir  = (float)gir[n0 + col];
            float iz  = (float)gir[512 + n0 + col];
            float inn = (float)gir[1024 + n0 + col];
            float hp  = (float)hprev[(size_t)row * H + n0 + col];
            float rg  = sigmoidf(ir + ar[i][r]);
            float zg  = sigmoidf(iz + az[i][r]);
            float ng  = tanhf(inn + rg * an[i][r]);
            float hv  = (1.0f - zg) * ng + zg * hp;
            hnext[(size_t)row * H + n0 + col]     = (_Float16)hv;
            encout[(size_t)row * 1024 + n0 + col] = (_Float16)hv;
        }
    }
}

// ---------------------------------------------------------------------------
// Persistent decoder GRU (scan over T=800, state = 32 x 512 kept in LDS
// double buffer). Single workgroup of 1024 threads = 32 waves. Wave w owns
// hidden columns [16w, 16w+16): both row-tiles (m0=0,16) share the three
// gate B-fragments, so Whh traffic is halved vs. one tile per wave. The
// next step's gi slab (96 KB) is prefetched (global_prefetch_b8) while the
// current step computes. Post projection fused via wave reductions.
// ---------------------------------------------------------------------------
__global__ void __launch_bounds__(1024)
gru_dec_run_kernel(const _Float16* __restrict__ gi,   // (800*32) x 1536
                   const _Float16* __restrict__ Whh,  // 1536 x 512
                   const float* __restrict__ bhh,     // 1536
                   const float* __restrict__ postW,   // 512
                   const float* __restrict__ postB,   // 1
                   float* __restrict__ out)           // (B,T) row-major b*800+t
{
    __shared__ __align__(16) _Float16 hbuf[2][32 * 512];

    int tid  = threadIdx.x;
    int lane = tid & 31;
    int wave = tid >> 5;
    int col = lane & 15, rh = lane >> 4;

    for (int i = tid; i < 32 * 512; i += 1024) hbuf[0][i] = (_Float16)0.0f;
    __syncthreads();

    int n0 = wave << 4;  // this wave's 16 hidden columns
    float br = bhh[n0 + col];
    float bz = bhh[512 + n0 + col];
    float bn = bhh[1024 + n0 + col];
    float pb = postB[0];

    for (int t = 0; t < 800; ++t) {
        const _Float16* src = hbuf[t & 1];
        _Float16* dst = hbuf[(t & 1) ^ 1];
        const _Float16* git = gi + (size_t)t * 32 * 1536;

        // Prefetch next step's gi slab: 49152 halves / 1024 threads = 48 each.
        if (t + 1 < 800) {
            const _Float16* gnext = git + 32 * 1536;
            __builtin_prefetch(gnext + tid * 48, 0, 1);
        }

        v8f ar[2], az[2], an[2];
#pragma unroll
        for (int i = 0; i < 2; ++i)
#pragma unroll
            for (int r = 0; r < 8; ++r) { ar[i][r] = br; az[i][r] = bz; an[i][r] = bn; }

        for (int k = 0; k < 512; k += 32) {
            v16h a0 = load_frag_a(src, 512, 0, k, lane);   // ds_load path
            v16h a1 = load_frag_a(src, 512, 16, k, lane);
            v16h b0 = load_frag_b(Whh, 512, n0, k, lane);
            v16h b1 = load_frag_b(Whh, 512, 512 + n0, k, lane);
            v16h b2 = load_frag_b(Whh, 512, 1024 + n0, k, lane);
            ar[0] = wmma_f16(a0, b0, ar[0]);
            az[0] = wmma_f16(a0, b1, az[0]);
            an[0] = wmma_f16(a0, b2, an[0]);
            ar[1] = wmma_f16(a1, b0, ar[1]);
            az[1] = wmma_f16(a1, b1, az[1]);
            an[1] = wmma_f16(a1, b2, an[1]);
        }

#pragma unroll
        for (int i = 0; i < 2; ++i) {
#pragma unroll
            for (int r = 0; r < 8; ++r) {
                int row = 16 * i + r + (rh << 3);
                const _Float16* gir = git + (size_t)row * 1536;
                float ir  = (float)gir[n0 + col];
                float iz  = (float)gir[512 + n0 + col];
                float inn = (float)gir[1024 + n0 + col];
                float hp  = (float)src[row * 512 + n0 + col];
                float rg  = sigmoidf(ir + ar[i][r]);
                float zg  = sigmoidf(iz + az[i][r]);
                float ng  = tanhf(inn + rg * an[i][r]);
                dst[row * 512 + n0 + col] = (_Float16)((1.0f - zg) * ng + zg * hp);
            }
        }
        __syncthreads();

        // Fused post projection: wave b computes out[b, t] = dot(h[b], postW)+pb
        const _Float16* hrow = dst + (wave << 9);
        float s = 0.0f;
#pragma unroll
        for (int i = 0; i < 16; ++i) {
            int j = lane + (i << 5);
            s += (float)hrow[j] * postW[j];
        }
#pragma unroll
        for (int off = 16; off; off >>= 1) s += __shfl_xor(s, off, 32);
        if (lane == 0) out[wave * 800 + t] = s + pb;
        // No trailing barrier needed: next step writes the *other* LDS buffer,
        // and all reads of `src` happened before the barrier above.
    }
}

// ---------------------------------------------------------------------------
// Host orchestration
// ---------------------------------------------------------------------------
extern "C" void kernel_launch(void* const* d_in, const int* in_sizes, int n_in,
                              void* d_out, int out_size, void* d_ws, size_t ws_size,
                              hipStream_t stream) {
    const int B = 32, T = 800, M = B * T;  // 25600

    const int*   note      = (const int*)d_in[0];
    const float* f0_prev   = (const float*)d_in[1];
    const float* rmse_prev = (const float*)d_in[2];
    const float* note_emb  = (const float*)d_in[3];
    const float* f0_W      = (const float*)d_in[4];
    const float* f0_b      = (const float*)d_in[5];
    const float* rm_W      = (const float*)d_in[6];
    const float* rm_b      = (const float*)d_in[7];
    const float* pre1_W    = (const float*)d_in[8];
    const float* pre1_b    = (const float*)d_in[9];
    const float* pre2_W    = (const float*)d_in[10];
    const float* pre2_b    = (const float*)d_in[11];
    const float* encf_Wih  = (const float*)d_in[12];
    const float* encf_Whh  = (const float*)d_in[13];
    const float* encf_bih  = (const float*)d_in[14];
    const float* encf_bhh  = (const float*)d_in[15];
    const float* encb_Wih  = (const float*)d_in[16];
    const float* encb_Whh  = (const float*)d_in[17];
    const float* encb_bih  = (const float*)d_in[18];
    const float* encb_bhh  = (const float*)d_in[19];
    const float* decf_Wih  = (const float*)d_in[20];
    const float* decf_Whh  = (const float*)d_in[21];
    const float* decf_bih  = (const float*)d_in[22];
    const float* decf_bhh  = (const float*)d_in[23];
    const float* decr_Wih  = (const float*)d_in[24];
    const float* decr_Whh  = (const float*)d_in[25];
    const float* decr_bih  = (const float*)d_in[26];
    const float* decr_bhh  = (const float*)d_in[27];
    const float* postf_W   = (const float*)d_in[28];
    const float* postf_b   = (const float*)d_in[29];
    const float* postr_W   = (const float*)d_in[30];
    const float* postr_b   = (const float*)d_in[31];

    float* out_f0 = (float*)d_out;
    float* out_rm = (float*)d_out + M;

    // Workspace carve (all f16 buffers, 256B aligned).
    size_t off = 0;
    auto alloc = [&](size_t nhalf) -> _Float16* {
        _Float16* p = (_Float16*)((char*)d_ws + off);
        off += ((nhalf * 2 + 255) & ~(size_t)255);
        return p;
    };
    _Float16* w_pre1    = alloc(256 * 256);
    _Float16* w_pre2    = alloc(256 * 256);
    _Float16* w_encf_ih = alloc(1536 * 256);
    _Float16* w_encf_hh = alloc(1536 * 512);
    _Float16* w_encb_ih = alloc(1536 * 256);
    _Float16* w_encb_hh = alloc(1536 * 512);
    _Float16* w_decf_ih = alloc(1536 * 1088);
    _Float16* w_decf_hh = alloc(1536 * 512);
    _Float16* w_decr_ih = alloc(1536 * 1088);
    _Float16* w_decr_hh = alloc(1536 * 512);
    _Float16* ne    = alloc((size_t)M * 256);
    _Float16* t0    = alloc((size_t)M * 256);
    _Float16* hpre  = alloc((size_t)M * 256);
    _Float16* gi_a  = alloc((size_t)M * 1536);  // encf, later decf
    _Float16* gi_b  = alloc((size_t)M * 1536);  // encb, later decr
    _Float16* enc   = alloc((size_t)M * 1024);
    _Float16* hA    = alloc((size_t)T * 512);
    _Float16* hB    = alloc((size_t)T * 512);
    _Float16* decin = alloc((size_t)M * 1088);

    auto cvt = [&](const float* s, _Float16* d, int n) {
        f32_to_f16_kernel<<<(n + 255) / 256, 256, 0, stream>>>(s, d, n);
    };
    cvt(pre1_W, w_pre1, 256 * 256);
    cvt(pre2_W, w_pre2, 256 * 256);
    cvt(encf_Wih, w_encf_ih, 1536 * 256);
    cvt(encf_Whh, w_encf_hh, 1536 * 512);
    cvt(encb_Wih, w_encb_ih, 1536 * 256);
    cvt(encb_Whh, w_encb_hh, 1536 * 512);
    cvt(decf_Wih, w_decf_ih, 1536 * 1088);
    cvt(decf_Whh, w_decf_hh, 1536 * 512);
    cvt(decr_Wih, w_decr_ih, 1536 * 1088);
    cvt(decr_Whh, w_decr_hh, 1536 * 512);

    // Embedding gather -> ne (M x 256) f16
    embed_gather_kernel<<<(M * 256 + 255) / 256, 256, 0, stream>>>(note, note_emb, ne, M);

    auto gemm = [&](const _Float16* X, const _Float16* Wm, const float* bias,
                    _Float16* Y, int Mm, int Nn, int Kk, bool do_tanh) {
        int blocks = ((Mm / 32) * (Nn / 64)) / 4;
        if (do_tanh)
            gemm_bias_act_kernel<1><<<blocks, 128, 0, stream>>>(X, Wm, bias, Y, Mm, Nn, Kk);
        else
            gemm_bias_act_kernel<0><<<blocks, 128, 0, stream>>>(X, Wm, bias, Y, Mm, Nn, Kk);
    };

    // Prenet
    gemm(ne, w_pre1, pre1_b, t0, M, 256, 256, true);
    gemm(t0, w_pre2, pre2_b, hpre, M, 256, 256, true);

    // Hoisted encoder input projections (bih folded in)
    gemm(hpre, w_encf_ih, encf_bih, gi_a, M, 1536, 256, false);
    gemm(hpre, w_encb_ih, encb_bih, gi_b, M, 1536, 256, false);

    // Encoder forward scan over B (32 steps), state (800 x 512)
    fill_zero_h_kernel<<<(T * 512 + 255) / 256, 256, 0, stream>>>(hA, T * 512);
    {
        _Float16 *ping = hA, *pong = hB;
        for (int s = 0; s < 32; ++s) {
            gru_enc_step_kernel<<<200, 128, 0, stream>>>(
                gi_a + (size_t)s * T * 1536, w_encf_hh, encf_bhh,
                ping, pong, enc + (size_t)s * T * 1024);
            _Float16* tmp = ping; ping = pong; pong = tmp;
        }
    }
    // Encoder backward scan: step s consumes b = 31-s, writes eb[b] at col 512
    fill_zero_h_kernel<<<(T * 512 + 255) / 256, 256, 0, stream>>>(hA, T * 512);
    {
        _Float16 *ping = hA, *pong = hB;
        for (int s = 0; s < 32; ++s) {
            int b = 31 - s;
            gru_enc_step_kernel<<<200, 128, 0, stream>>>(
                gi_b + (size_t)b * T * 1536, w_encb_hh, encb_bhh,
                ping, pong, enc + (size_t)b * T * 1024 + 512);
            _Float16* tmp = ping; ping = pong; pong = tmp;
        }
    }

    // Decoder F0: build (T,B) scan-major input, hoist gi, persistent scan
    build_decin_kernel<<<(int)((25600LL * 1088 + 255) / 256), 256, 0, stream>>>(
        enc, f0_prev, f0_W, f0_b, decin);
    gemm(decin, w_decf_ih, decf_bih, gi_a, M, 1536, 1088, false);
    gru_dec_run_kernel<<<1, 1024, 0, stream>>>(gi_a, w_decf_hh, decf_bhh,
                                               postf_W, postf_b, out_f0);

    // Decoder RMSE
    build_decin_kernel<<<(int)((25600LL * 1088 + 255) / 256), 256, 0, stream>>>(
        enc, rmse_prev, rm_W, rm_b, decin);
    gemm(decin, w_decr_ih, decr_bih, gi_b, M, 1536, 1088, false);
    gru_dec_run_kernel<<<1, 1024, 0, stream>>>(gi_b, w_decr_hh, decr_bhh,
                                               postr_W, postr_b, out_rm);

    (void)in_sizes; (void)n_in; (void)out_size; (void)ws_size;
}
